// DynamicPositioningAttention3D_53291954209155
// MI455X (gfx1250) — matrix operand order repaired and checked by reading, not compiled, not run on hardware
//
#include <hip/hip_runtime.h>

// Problem constants from the reference: B,C,D,H,W = 2,32,64,96,96
#define Bn 2
#define Cn 32
#define Dn 64
#define Hn 96
#define Wn 96
#define HWn (Hn * Wn)          // 9216
#define Sn  (Dn * Hn * Wn)     // 589824 spatial elems per (b,c) plane
#define CHUNKS 16              // partial-sum chunks per plane for mean(x)
#define CP 40                  // padded channel stride (bf16/bytes): 80B rows
#define OP 36                  // padded channel stride for f32 out staging

typedef __attribute__((ext_vector_type(8)))  __bf16 v8bf;
typedef __attribute__((ext_vector_type(16))) __bf16 v16bf;
typedef __attribute__((ext_vector_type(8)))  float  v8f;

// ---------------------------------------------------------------------------
// WMMA fragment builders per CDNA5 ISA 7.12.2 (wave32, 16x16x32 bf16).
// A (16x32 MxK): lane half h, m=l15; elems 0..7 -> K=h*8..+7,
//                elems 8..15 -> K=16+h*8..+7 (two aligned 16B chunks)
// B (32x16 KxN): lane half h, n=l15; same K order as A (two aligned 16B
//                chunks when X is stored transposed [n][k])
// C/D (16x16):   lane half h, n=l15; elem r -> M = r + 8*h
// ---------------------------------------------------------------------------
__device__ __forceinline__ v16bf load_A_frag(const __bf16 (*W)[Cn], int mt,
                                             int half, int l15) {
  const __bf16* p = &W[mt * 16 + l15][half * 8];
  v8bf lo = *(const v8bf*)p;
  v8bf hi = *(const v8bf*)(p + 16);
  return __builtin_shufflevector(lo, hi, 0, 1, 2, 3, 4, 5, 6, 7, 8, 9, 10, 11,
                                 12, 13, 14, 15);
}

__device__ __forceinline__ v16bf load_B_fragT(const __bf16 (*XT)[CP], int nt,
                                              int half, int l15) {
  const __bf16* p = &XT[nt * 16 + l15][half * 8];  // 16B aligned (80B rows)
  v8bf lo = *(const v8bf*)p;
  v8bf hi = *(const v8bf*)(p + 16);
  return __builtin_shufflevector(lo, hi, 0, 1, 2, 3, 4, 5, 6, 7, 8, 9, 10, 11,
                                 12, 13, 14, 15);
}

__device__ __forceinline__ v8f wmma_bf16(v16bf a, v16bf b, v8f c) {
  return __builtin_amdgcn_wmma_f32_16x16x32_bf16(false, a, false, b,
                                                 (short)0, c, false, false);
}

// ---------------------------------------------------------------------------
// Pass 1: per-(b,c) partial sums of x (deterministic: no float atomics).
// ---------------------------------------------------------------------------
__global__ __launch_bounds__(256) void xsum_partial_kernel(
    const float* __restrict__ x, float* __restrict__ partial) {
  const int plane = blockIdx.x / CHUNKS;   // b*Cn + c
  const int chunk = blockIdx.x % CHUNKS;
  const int n4 = (Sn / CHUNKS) / 4;
  const float4* p =
      (const float4*)(x + (size_t)plane * Sn + (size_t)chunk * (Sn / CHUNKS));
  float acc = 0.f;
  for (int t = threadIdx.x; t < n4; t += 256) {
    float4 v = p[t];
    acc += (v.x + v.y) + (v.z + v.w);
  }
  __shared__ float red[256];
  red[threadIdx.x] = acc;
  __syncthreads();
  for (int s = 128; s > 0; s >>= 1) {
    if (threadIdx.x < s) red[threadIdx.x] += red[threadIdx.x + s];
    __syncthreads();
  }
  if (threadIdx.x == 0) partial[blockIdx.x] = red[0];
}

__global__ void xsum_reduce_kernel(const float* __restrict__ partial,
                                   float* __restrict__ xsum) {
  const int t = threadIdx.x;  // 0..63
  float a = 0.f;
#pragma unroll
  for (int k = 0; k < CHUNKS; ++k) a += partial[t * CHUNKS + k];
  xsum[t] = a;
}

// Stage one x row (f32 global float4 loads) into transposed bf16 LDS tile.
__device__ __forceinline__ void stage_row(const float* __restrict__ xrow,
                                          __bf16 (*xtT)[CP], int tid) {
  for (int i = tid; i < Cn * (Wn / 4); i += 128) {
    const int c = i / (Wn / 4), x4 = (i % (Wn / 4)) << 2;
    float4 f = *(const float4*)(xrow + (size_t)c * Sn + x4);
    xtT[x4 + 0][c] = (__bf16)f.x;
    xtT[x4 + 1][c] = (__bf16)f.y;
    xtT[x4 + 2][c] = (__bf16)f.z;
    xtT[x4 + 3][c] = (__bf16)f.w;
  }
}

// Neighbor row: Q GEMM only, pack A_S compare bits -> one ds_store_b64/tile.
__device__ __forceinline__ void q_row_neighbor(
    const float* __restrict__ xrow, __bf16 (*xtT)[CP],
    unsigned char (*asbR)[CP], v16bf aQ, v8f bq8, v8f avg8, int tid, int wave,
    int half, int l15, int m0) {
  stage_row(xrow, xtT, tid);
  __syncthreads();
#pragma unroll
  for (int i = 0; i < 3; ++i) {
    const int nt = (wave & 1) * 3 + i;
    v16bf bb = load_B_fragT(xtT, nt, half, l15);
    v8f acc = {};
    acc = wmma_bf16(aQ, bb, acc);
    const int n = nt * 16 + l15;
    unsigned long long bits = 0ull;
#pragma unroll
    for (int rr = 0; rr < 8; ++rr) {
      const float v = acc[rr] + bq8[rr];
      bits |= (unsigned long long)(v >= avg8[rr] ? 1u : 0u) << (8 * rr);
    }
    *(unsigned long long*)&asbR[n][m0] = bits;
  }
  __syncthreads();
}

// ---------------------------------------------------------------------------
// Fused row kernel: one block per (b,z,y) row of 96 voxels, 128 thr = 4 waves.
// Wave w owns m-tile (w/2) and n-tiles (w&1)*3..+2 -> per-wave-constant A
// fragments and bias/avg octets hoisted to registers.
// ---------------------------------------------------------------------------
__global__ __launch_bounds__(128) void dpa3d_row_kernel(
    const float* __restrict__ x,
    const float* __restrict__ wq, const float* __restrict__ bq,
    const float* __restrict__ wk, const float* __restrict__ bk,
    const float* __restrict__ wv, const float* __restrict__ bv,
    const float* __restrict__ wo, const float* __restrict__ bo,
    const float* __restrict__ xsum, float* __restrict__ out) {
  const int rowid = blockIdx.x;  // b*Dn*Hn + z*Hn + y
  const int y = rowid % Hn;
  const int z = (rowid / Hn) % Dn;
  const int b = rowid / (Dn * Hn);

  const int tid  = threadIdx.x;
  const int wave = tid >> 5;
  const int lane = tid & 31;
  const int half = lane >> 4;
  const int l15  = lane & 15;
  const int mt   = wave >> 1;
  const int m0   = mt * 16 + half * 8;  // per-lane constant output row base

  __shared__ __bf16 wbf[4][Cn][Cn];  // wq,wk,wv,wo (bf16)
  __shared__ float  biasL[4][Cn];
  __shared__ float  avgQ[Cn];
  __shared__ __bf16 xtT[Wn][CP];     // one x row, transposed [x][c]
  __shared__ union {                 // asb dead before oT is born
    unsigned char asb[5][Wn][CP];    // A_S bytes, transposed [x][c]
    float oT[Wn][OP];                // staged output, transposed [x][c]
  } u;
  __shared__ __bf16 qcT[Wn][CP];
  __shared__ __bf16 kcT[Wn][CP];
  __shared__ __bf16 vT[Wn][CP];      // V, then V*win in place
  __shared__ unsigned char winT[Wn][CP];
  __shared__ float sc[Wn];
  __shared__ float pr[Wn];

  // ---- stage weights/biases; analytic avg_Q -------------------------------
  for (int i = tid; i < Cn * Cn; i += 128) {
    ((__bf16*)wbf[0])[i] = (__bf16)wq[i];
    ((__bf16*)wbf[1])[i] = (__bf16)wk[i];
    ((__bf16*)wbf[2])[i] = (__bf16)wv[i];
    ((__bf16*)wbf[3])[i] = (__bf16)wo[i];
  }
  if (tid < Cn) {
    biasL[0][tid] = bq[tid]; biasL[1][tid] = bk[tid];
    biasL[2][tid] = bv[tid]; biasL[3][tid] = bo[tid];
    float a = 0.f;
#pragma unroll
    for (int c = 0; c < Cn; ++c) a += wq[tid * Cn + c] * xsum[b * Cn + c];
    avgQ[tid] = a * (1.0f / (float)Sn) + bq[tid];
  }
  if (tid < Cn) {  // prefetch central row, one line per channel
    __builtin_prefetch(
        x + ((size_t)(b * Cn + tid) * Dn + z) * HWn + (size_t)y * Wn, 0, 0);
  }
  __syncthreads();

  // ---- hoist per-wave A fragments + per-lane bias/avg octets --------------
  const v16bf aQ = load_A_frag(wbf[0], mt, half, l15);
  const v16bf aK = load_A_frag(wbf[1], mt, half, l15);
  const v16bf aV = load_A_frag(wbf[2], mt, half, l15);
  const v16bf aO = load_A_frag(wbf[3], mt, half, l15);
  v8f bq8, bk8, bv8, bo8, avg8;
#pragma unroll
  for (int rr = 0; rr < 8; ++rr) {
    bq8[rr] = biasL[0][m0 + rr];
    bk8[rr] = biasL[1][m0 + rr];
    bv8[rr] = biasL[2][m0 + rr];
    bo8[rr] = biasL[3][m0 + rr];
    avg8[rr] = avgQ[m0 + rr];
  }

  // ---- A_S for the 4 Laplacian neighbor rows ------------------------------
  const float* xbase = x + (size_t)(b * Cn) * Dn * HWn;
#define DO_NROW(Ridx, ZZ, YY)                                                  \
  {                                                                            \
    const int zz = (ZZ), yy = (YY);                                            \
    if (zz >= 0 && zz < Dn && yy >= 0 && yy < Hn) {                            \
      q_row_neighbor(xbase + (size_t)zz * HWn + (size_t)yy * Wn, xtT,          \
                     u.asb[Ridx], aQ, bq8, avg8, tid, wave, half, l15, m0);    \
    } else {                                                                   \
      for (int i = tid; i < (Wn * CP) / 16; i += 128)                          \
        ((uint4*)u.asb[Ridx])[i] = make_uint4(0u, 0u, 0u, 0u);                 \
      __syncthreads();                                                         \
    }                                                                          \
  }
  DO_NROW(0, z - 1, y)
  DO_NROW(1, z - 1, y + 1)
  DO_NROW(3, z + 1, y)
  DO_NROW(4, z + 1, y + 1)
#undef DO_NROW

  // ---- central row: shared B fragment -> Q, K, V WMMAs back-to-back -------
  stage_row(xbase + (size_t)z * HWn + (size_t)y * Wn, xtT, tid);
  __syncthreads();
#pragma unroll
  for (int i = 0; i < 3; ++i) {
    const int nt = (wave & 1) * 3 + i;
    v16bf bb = load_B_fragT(xtT, nt, half, l15);
    v8f accQ = {}, accK = {}, accV = {};
    accQ = wmma_bf16(aQ, bb, accQ);   // independent D's: no hazard NOPs,
    accK = wmma_bf16(aK, bb, accK);   // B operand shared across all three
    accV = wmma_bf16(aV, bb, accV);
    const int n = nt * 16 + l15;
    unsigned long long bits = 0ull;
    v8bf q8, k8, v8;
#pragma unroll
    for (int rr = 0; rr < 8; ++rr) {
      const float qv = accQ[rr] + bq8[rr];
      bits |= (unsigned long long)(qv >= avg8[rr] ? 1u : 0u) << (8 * rr);
      q8[rr] = (__bf16)qv;
      k8[rr] = (__bf16)(accK[rr] + bk8[rr]);
      v8[rr] = (__bf16)(accV[rr] + bv8[rr]);
    }
    *(unsigned long long*)&u.asb[2][n][m0] = bits;
    *(v8bf*)&qcT[n][m0] = q8;
    *(v8bf*)&kcT[n][m0] = k8;
    *(v8bf*)&vT[n][m0]  = v8;
  }
  __syncthreads();

  // ---- Laplacian window + masked Q·K scores, 4 channels per u32 -----------
  // A_S bytes are 0/1: pos = 4*A2 + A1 + A4 (<=6), neg = A2[x-1]+A2[x+1]
  // + A0 + A3 (<=4); per-byte, carry-free. win = pos > neg via borrow trick.
  if (tid < Wn) {
    const int xx = tid;
    const int xm = (xx > 0) ? xx - 1 : xx;
    const int xp = (xx < Wn - 1) ? xx + 1 : xx;
    const unsigned int mskm = (xx > 0) ? 0xFFFFFFFFu : 0u;
    const unsigned int mskp = (xx < Wn - 1) ? 0xFFFFFFFFu : 0u;
    float s = 0.f;
#pragma unroll
    for (int c4 = 0; c4 < Cn; c4 += 4) {
      const unsigned int a0 = *(const unsigned int*)&u.asb[0][xx][c4];
      const unsigned int a1 = *(const unsigned int*)&u.asb[1][xx][c4];
      const unsigned int a2 = *(const unsigned int*)&u.asb[2][xx][c4];
      const unsigned int a3 = *(const unsigned int*)&u.asb[3][xx][c4];
      const unsigned int a4 = *(const unsigned int*)&u.asb[4][xx][c4];
      const unsigned int am = (*(const unsigned int*)&u.asb[2][xm][c4]) & mskm;
      const unsigned int ap = (*(const unsigned int*)&u.asb[2][xp][c4]) & mskp;
      const unsigned int pos = (a2 << 2) + a1 + a4;        // <= 6 per byte
      const unsigned int neg = am + ap + a0 + a3;          // <= 4 per byte
      const unsigned int gt =
          (((pos | 0x80808080u) - neg - 0x01010101u) >> 7) & 0x01010101u;
      *(unsigned int*)&winT[xx][c4] = gt;                  // 0/1 bytes
#pragma unroll
      for (int j = 0; j < 4; ++j) {
        const float wf = (float)((gt >> (8 * j)) & 1u);
        s += wf * (float)qcT[xx][c4 + j] * (float)kcT[xx][c4 + j];
      }
    }
    sc[xx] = s * 0.17677669529663687f;  // 1/sqrt(32)
  }
  __syncthreads();

  // ---- softmax over the W axis (96), float4-vectorized reductions ---------
  float mx = -3.0e38f;
  for (int i = 0; i < Wn / 4; ++i) {
    float4 v = ((const float4*)sc)[i];
    mx = fmaxf(mx, fmaxf(fmaxf(v.x, v.y), fmaxf(v.z, v.w)));
  }
  if (tid < Wn) pr[tid] = __expf(sc[tid] - mx);
  __syncthreads();
  float sum = 0.f;
  for (int i = 0; i < Wn / 4; ++i) {
    float4 v = ((const float4*)pr)[i];
    sum += (v.x + v.y) + (v.z + v.w);
  }
  const float invsum = 1.0f / sum;

  // ---- mask V in place ----------------------------------------------------
  for (int i = tid; i < Wn * Cn; i += 128) {
    const int xx = i >> 5, c = i & 31;
    if (!winT[xx][c]) vT[xx][c] = (__bf16)0.f;
  }
  __syncthreads();

  // ---- output GEMM: out = attw * (Wo x Vd) + bo ---------------------------
#pragma unroll
  for (int i = 0; i < 3; ++i) {
    const int nt = (wave & 1) * 3 + i;
    v16bf bb = load_B_fragT(vT, nt, half, l15);
    v8f acc = {};
    acc = wmma_bf16(aO, bb, acc);
    const int n = nt * 16 + l15;
    const float att = pr[n] * invsum;
    float4 lo4, hi4;
    lo4.x = att * acc[0] + bo8[0];
    lo4.y = att * acc[1] + bo8[1];
    lo4.z = att * acc[2] + bo8[2];
    lo4.w = att * acc[3] + bo8[3];
    hi4.x = att * acc[4] + bo8[4];
    hi4.y = att * acc[5] + bo8[5];
    hi4.z = att * acc[6] + bo8[6];
    hi4.w = att * acc[7] + bo8[7];
    *(float4*)&u.oT[n][m0]     = lo4;  // 16B-aligned (OP=36 -> 144B rows)
    *(float4*)&u.oT[n][m0 + 4] = hi4;
  }
  __syncthreads();

  // ---- coalesced float4 row store -----------------------------------------
  float* orow = out + ((size_t)(b * Cn) * Dn + z) * HWn + (size_t)y * Wn;
  for (int i = tid; i < Cn * (Wn / 4); i += 128) {
    const int c = i / (Wn / 4), x4 = (i % (Wn / 4)) << 2;
    float4 f;
    f.x = u.oT[x4 + 0][c];
    f.y = u.oT[x4 + 1][c];
    f.z = u.oT[x4 + 2][c];
    f.w = u.oT[x4 + 3][c];
    *(float4*)(orow + (size_t)c * Sn + x4) = f;
  }
}

// ---------------------------------------------------------------------------
extern "C" void kernel_launch(void* const* d_in, const int* in_sizes, int n_in,
                              void* d_out, int out_size, void* d_ws,
                              size_t ws_size, hipStream_t stream) {
  (void)in_sizes; (void)n_in; (void)out_size; (void)ws_size;
  const float* x  = (const float*)d_in[0];
  const float* wq = (const float*)d_in[1];
  const float* bq = (const float*)d_in[2];
  const float* wk = (const float*)d_in[3];
  const float* bk = (const float*)d_in[4];
  const float* wv = (const float*)d_in[5];
  const float* bv = (const float*)d_in[6];
  const float* wo = (const float*)d_in[7];
  const float* bo = (const float*)d_in[8];
  float* out = (float*)d_out;

  float* xsum    = (float*)d_ws;        // 64 floats
  float* partial = (float*)d_ws + 64;   // Bn*Cn*CHUNKS floats

  xsum_partial_kernel<<<Bn * Cn * CHUNKS, 256, 0, stream>>>(x, partial);
  xsum_reduce_kernel<<<1, Bn * Cn, 0, stream>>>(partial, xsum);
  dpa3d_row_kernel<<<Bn * Dn * Hn, 128, 0, stream>>>(
      x, wq, bq, wk, bk, wv, bv, wo, bo, xsum, out);
}
